// StyleEncoder_27891517620746
// MI455X (gfx1250) — matrix operand order repaired and checked by reading, not compiled, not run on hardware
//
#include <hip/hip_runtime.h>
#include <hip/hip_bf16.h>
#include <math.h>
#include <stdint.h>

// ---------------------------------------------------------------------------
// Problem constants (fixed by setup_inputs)
// ---------------------------------------------------------------------------
#define BATCH 16
#define TLEN  2048
#define LLEN  512
#define HC    256      // hidden
#define H2    512      // 2*hidden
#define CVQ_C 64       // vq channel
#define KCODE 128      // codebook entries
#define NLAY  5
#define NBLK  2

typedef __attribute__((ext_vector_type(16))) _Float16 v16h;
typedef __attribute__((ext_vector_type(8)))  _Float16 v8h;
typedef __attribute__((ext_vector_type(8)))  float    v8f;
typedef __attribute__((ext_vector_type(4)))  unsigned int v4u;
typedef __attribute__((ext_vector_type(8)))  int      v8i;
typedef __attribute__((ext_vector_type(4)))  int      v4i;

#define TILE_M 64
#define TILE_N 64
#define TILE_K 32
#define A_STRIDE 36                   // 32 dwords + 4 dword TDM pad (bank skew, 16B align)
#define LDS_STRIDE (TILE_K + 8)       // 40 halves = 80 B (16B-aligned rows, bank skew)
#define GEMM_THREADS 128              // 4 wave32 waves

__device__ __forceinline__ float gelu_exact(float v) {
    return 0.5f * v * (1.0f + erff(v * 0.70710678118654752f));
}

// ---------------------------------------------------------------------------
// TDM: async DMA of a 2D f32 tile [64 rows x 32 cols] from global into LDS.
// D# built per CDNA5 ISA 8.3-8.6. tensor_dim0/1 = remaining cols/rows so the
// hardware zero-fills out-of-bounds reads (replaces K-tail guards).
// LDS padding: +4 dwords every 32 dwords -> row stride 36 dwords.
// This toolchain (clang-23 / therock-10.0) uses the 6-arg builtin:
//   (v4u g0, v8i g1, v4i g2, v4i g3, v8i g4, i32 cpol)
// ---------------------------------------------------------------------------
__device__ __forceinline__ void tdm_load_a_tile(unsigned lds_byte_addr,
                                                const float* gptr,
                                                unsigned rem_cols,
                                                unsigned rem_rows,
                                                unsigned pitch_elems)
{
    unsigned long long ga = (unsigned long long)(uintptr_t)gptr;
    v4u g0;
    g0[0] = 1u;                                            // count=1 (valid user D#)
    g0[1] = lds_byte_addr;                                 // lds_addr (bytes)
    g0[2] = (unsigned)(ga & 0xFFFFFFFFu);                  // global_addr[31:0]
    g0[3] = (unsigned)((ga >> 32) & 0x01FFFFFFu) | 0x80000000u;  // ga[56:32] | type=2
    v8i g1;
    // data_size=2 (4B) | pad_enable | pad_interval=4 (32 dwords) | pad_amount=3 (4 dwords)
    g1[0] = (int)0x07120000u;
    g1[1] = (int)((rem_cols & 0xFFFFu) << 16);             // tensor_dim0[15:0]
    g1[2] = (int)(((rem_cols >> 16) & 0xFFFFu) | ((rem_rows & 0xFFFFu) << 16));
    g1[3] = (int)(((rem_rows >> 16) & 0xFFFFu) | ((unsigned)TILE_K << 16)); // tile_dim0=32
    g1[4] = (int)TILE_M;                                   // tile_dim1=64, tile_dim2=0
    g1[5] = (int)pitch_elems;                              // tensor_dim0_stride[31:0]
    g1[6] = 0;
    g1[7] = 0;
    v4i g2 = {0, 0, 0, 0};                                 // 2D tensor: dims 2/3 unused
    v4i g3 = {0, 0, 0, 0};
    v8i g4 = {0, 0, 0, 0, 0, 0, 0, 0};
    __builtin_amdgcn_tensor_load_to_lds(g0, g1, g2, g3, g4, 0);
}

// ---------------------------------------------------------------------------
// WMMA GEMM: Y[b,Cout,T] = epilogue( W[Cout,Cin*KSIZE] @ im2col(X[b,Cin,T]) )
// KSIZE=1 -> pointwise conv; KSIZE=3 -> causal conv (left pad KSIZE-1).
// Epilogue: v = (acc + bias) * scale; [gelu]; [+resid]; [*mask]
// A tiles arrive via double-buffered tensor_load_to_lds (TENSORcnt pipelined);
// B tiles staged K-contiguous f16 so each fragment = 2x ds_load_b128.
// NOTE: launches guarantee Cout % 64 == 0 and T % 64 == 0 (no epilogue guards).
// ---------------------------------------------------------------------------
template<int KSIZE, bool GELU_ON>
__global__ __launch_bounds__(GEMM_THREADS)
void wmma_conv_gemm(const float* __restrict__ X,     // [B, Cin, T]
                    const float* __restrict__ W,     // [Cout, Cin, KSIZE]
                    const float* __restrict__ bias,  // [Cout] or null
                    const float* __restrict__ resid, // [B, Cout, T] or null
                    const float* __restrict__ mask,  // [B, T] or null
                    float* __restrict__ Y,           // [B, Cout, T]
                    int Cin, int Cout, int T, float scale)
{
    __shared__ __align__(16) float    Af32[2][TILE_M][A_STRIDE]; // TDM-filled (row pad 4)
    __shared__ __align__(16) _Float16 BsT[TILE_N][LDS_STRIDE];   // [n][k] (transposed)

    const int b    = blockIdx.z;
    const int n0   = blockIdx.x * TILE_N;   // time tile
    const int m0   = blockIdx.y * TILE_M;   // out-channel tile
    const int tid  = threadIdx.x;
    const int lane = tid & 31;
    const int wave = tid >> 5;
    const int wm   = (wave >> 1) * 32;      // wave sub-tile within block tile
    const int wn   = (wave & 1) * 32;
    const int rh   = lane & 15;             // index within 16-half
    const int hi   = lane >> 4;             // half-wave select

    // B staging: each thread owns row stg_r, 16 K-contiguous elements
    const int stg_r = tid >> 1;             // 0..63
    const int stg_k = (tid & 1) * 16;       // 0 or 16

    const float* __restrict__ Xb = X + (size_t)b * Cin * T;
    const int Kdim = Cin * KSIZE;

    const unsigned ldsA0 = (unsigned)(uintptr_t)(void*)&Af32[0][0][0];
    const unsigned ldsA1 = (unsigned)(uintptr_t)(void*)&Af32[1][0][0];

    // prologue: kick off DMA of the first A tile
    if (wave == 0) {
        tdm_load_a_tile(ldsA0, W + (size_t)m0 * Kdim,
                        (unsigned)Kdim, (unsigned)(Cout - m0), (unsigned)Kdim);
    }

    v8f acc[2][2] = {};
    int cur = 0;

    #pragma unroll 1
    for (int k0 = 0; k0 < Kdim; k0 += TILE_K) {
        const bool more = (k0 + TILE_K) < Kdim;
        const int  kbase = k0 + stg_k;

        // ---- stage B transposed (causal im2col gather; coalesced along n) ----
        {
            const int n = stg_r;
            v8h lo = {}, hi8 = {};
            if (KSIZE == 1) {
                const int t = n0 + n;                    // always in [0, T)
                #pragma unroll
                for (int u = 0; u < 16; ++u) {
                    int k = kbase + u;
                    float v = (k < Kdim) ? Xb[(size_t)k * T + t] : 0.0f;
                    if (u < 8) lo[u] = (_Float16)v; else hi8[u - 8] = (_Float16)v;
                }
            } else {
                int cin = kbase / KSIZE;                 // one division per K-step
                int kk  = kbase - cin * KSIZE;
                #pragma unroll
                for (int u = 0; u < 16; ++u) {
                    float v = 0.0f;
                    int t = n0 + n + kk - (KSIZE - 1);
                    if ((kbase + u) < Kdim && t >= 0 && t < T)
                        v = Xb[(size_t)cin * T + t];
                    if (u < 8) lo[u] = (_Float16)v; else hi8[u - 8] = (_Float16)v;
                    if (++kk == KSIZE) { kk = 0; ++cin; }
                }
            }
            *(v8h*)&BsT[n][stg_k]     = lo;
            *(v8h*)&BsT[n][stg_k + 8] = hi8;
        }
        // speculative prefetch of next iteration's X rows
        if (more) {
            int kx = kbase + TILE_K;
            if (kx < Kdim) {
                int cinx = kx / KSIZE;
                __builtin_prefetch(Xb + (size_t)cinx * T + n0 + stg_r, 0, 0);
            }
        }

        // ---- async-tensor pipeline: issue next A tile, wait for current ----
        if (wave == 0) {
            if (more) {
                tdm_load_a_tile(cur ? ldsA0 : ldsA1,
                                W + (size_t)m0 * Kdim + (k0 + TILE_K),
                                (unsigned)(Kdim - (k0 + TILE_K)),
                                (unsigned)(Cout - m0), (unsigned)Kdim);
                __builtin_amdgcn_s_wait_tensorcnt(1);   // current tile done
            } else {
                __builtin_amdgcn_s_wait_tensorcnt(0);   // last tile done
            }
        }
        __syncthreads();

        // ---- fragments (ISA 7.12.2 wave32 layouts) ----
        v16h afrag[2], bfrag[2];
        #pragma unroll
        for (int mi = 0; mi < 2; ++mi) {
            int r = wm + mi * 16 + rh;
            const float* ap = &Af32[cur][r][0];
            #pragma unroll
            for (int u = 0; u < 8; ++u) {
                afrag[mi][u]     = (_Float16)ap[hi * 8 + u];       // K 0..7   / 8..15
                afrag[mi][8 + u] = (_Float16)ap[16 + hi * 8 + u];  // K 16..23 / 24..31
            }
        }
        #pragma unroll
        for (int ni = 0; ni < 2; ++ni) {
            int c = wn + ni * 16 + rh;
            v8h l0 = *(const v8h*)&BsT[c][hi * 8];       // K 0..7   / 8..15
            v8h l1 = *(const v8h*)&BsT[c][16 + hi * 8];  // K 16..23 / 24..31
            bfrag[ni] = __builtin_shufflevector(l0, l1, 0,1,2,3,4,5,6,7,8,9,10,11,12,13,14,15);
        }
        #pragma unroll
        for (int mi = 0; mi < 2; ++mi)
            #pragma unroll
            for (int ni = 0; ni < 2; ++ni)
                acc[mi][ni] = __builtin_amdgcn_wmma_f32_16x16x32_f16(
                    false, afrag[mi], false, bfrag[ni],
                    (short)0, acc[mi][ni], false, false);
        __syncthreads();
        cur ^= 1;
    }

    // ---- epilogue (shapes are multiples of the tile; no bounds guards) ----
    const float* residb = resid ? resid + (size_t)b * Cout * T : (const float*)0;
    const float* maskb  = mask  ? mask  + (size_t)b * T        : (const float*)0;
    float* Yb = Y + (size_t)b * Cout * T;
    #pragma unroll
    for (int mi = 0; mi < 2; ++mi)
        #pragma unroll
        for (int ni = 0; ni < 2; ++ni) {
            const int n = n0 + wn + ni * 16 + rh;        // N = lane&15
            const float mk = maskb ? maskb[n] : 1.0f;
            #pragma unroll
            for (int i = 0; i < 8; ++i) {
                const int m = m0 + wm + mi * 16 + hi * 8 + i;  // C/D: M = i + 8*hi
                float v = acc[mi][ni][i];
                if (bias) v += bias[m];
                v *= scale;
                if (GELU_ON) v = gelu_exact(v);
                if (residb) v += residb[(size_t)m * T + n];
                v *= mk;
                Yb[(size_t)m * T + n] = v;
            }
        }
}

// ---------------------------------------------------------------------------
// Channel LayerNorm over C=256 of [B,C,T]; one wave per (b,t) column.
// ---------------------------------------------------------------------------
__global__ __launch_bounds__(256)
void ln_kernel(const float* __restrict__ X, const float* __restrict__ g,
               const float* __restrict__ bta, const float* __restrict__ mask,
               float* __restrict__ Y, int T, int ncols)
{
    int wv = threadIdx.x >> 5, lane = threadIdx.x & 31;
    int col = blockIdx.x * 8 + wv;
    if (col >= ncols) return;
    int b = col / T, t = col % T;
    const float* xp = X + (size_t)b * HC * T + t;
    float v[8];
    float s = 0.0f;
    #pragma unroll
    for (int i = 0; i < 8; ++i) { v[i] = xp[(size_t)(lane + 32 * i) * T]; s += v[i]; }
    #pragma unroll
    for (int off = 16; off > 0; off >>= 1) s += __shfl_xor(s, off, 32);
    float mean = s * (1.0f / 256.0f);
    float q = 0.0f;
    #pragma unroll
    for (int i = 0; i < 8; ++i) { float d = v[i] - mean; q += d * d; }
    #pragma unroll
    for (int off = 16; off > 0; off >>= 1) q += __shfl_xor(q, off, 32);
    float rstd = rsqrtf(q * (1.0f / 256.0f) + 1e-5f);
    float mk = mask ? mask[(size_t)b * T + t] : 1.0f;
    float* yp = Y + (size_t)b * HC * T + t;
    #pragma unroll
    for (int i = 0; i < 8; ++i) {
        int c = lane + 32 * i;
        yp[(size_t)c * T] = ((v[i] - mean) * rstd * g[c] + bta[c]) * mk;
    }
}

// ---------------------------------------------------------------------------
// group_by_segs helpers
// ---------------------------------------------------------------------------
__global__ void zero_kernel(float* p, size_t n) {
    size_t i = (size_t)blockIdx.x * blockDim.x + threadIdx.x;
    size_t st = (size_t)gridDim.x * blockDim.x;
    for (; i < n; i += st) p[i] = 0.0f;
}

__global__ __launch_bounds__(256)
void scatter_kernel(const float* __restrict__ h, const int* __restrict__ mel2ph,
                    float* __restrict__ sums, float* __restrict__ counts)
{
    int b = blockIdx.y;
    int i = blockIdx.x * 256 + threadIdx.x;          // over T*H
    if (i >= TLEN * HC) return;
    int t = i / HC, c = i % HC;
    int m = mel2ph[b * TLEN + t];                    // 1..L
    atomicAdd(&sums[((size_t)b * HC + c) * (LLEN + 1) + m],
              h[((size_t)b * HC + c) * TLEN + t]);
    if (c == 0) atomicAdd(&counts[b * (LLEN + 1) + m], 1.0f);
}

__global__ __launch_bounds__(256)
void group_norm_kernel(const float* __restrict__ sums, const float* __restrict__ counts,
                       float* __restrict__ g)
{
    int b = blockIdx.y;
    int i = blockIdx.x * 256 + threadIdx.x;          // over H*L
    if (i >= HC * LLEN) return;
    int c = i / LLEN, l = i % LLEN;
    float cnt = counts[b * (LLEN + 1) + l + 1];
    g[((size_t)b * HC + c) * LLEN + l] =
        sums[((size_t)b * HC + c) * (LLEN + 1) + l + 1] / fmaxf(cnt, 1.0f);
}

// ---------------------------------------------------------------------------
// VQ: one block per (b,l); thread k computes ||x - c_k||^2 (same argmin as
// the expanded-form distance), block argmin, q gather, commitment-loss accum.
// ---------------------------------------------------------------------------
__global__ __launch_bounds__(KCODE)
void vq_kernel(const float* __restrict__ z, const float* __restrict__ codebook,
               float* __restrict__ q, int* __restrict__ idx,
               float* __restrict__ loss_accum)
{
    int bl = blockIdx.x;
    int b = bl / LLEN, l = bl % LLEN;
    __shared__ float xs[CVQ_C];
    __shared__ float dist[KCODE];
    __shared__ int kmin_s;
    const float* zp = z + (size_t)b * CVQ_C * LLEN + l;
    if (threadIdx.x < CVQ_C) xs[threadIdx.x] = zp[(size_t)threadIdx.x * LLEN];
    __syncthreads();
    int k = threadIdx.x;
    const float* ck = codebook + (size_t)k * CVQ_C;
    float d = 0.0f;
    #pragma unroll 8
    for (int c = 0; c < CVQ_C; ++c) { float df = xs[c] - ck[c]; d += df * df; }
    dist[k] = d;
    __syncthreads();
    if (threadIdx.x == 0) {
        float best = dist[0]; int bi = 0;
        for (int i = 1; i < KCODE; ++i)
            if (dist[i] < best) { best = dist[i]; bi = i; }
        kmin_s = bi;
        idx[b * LLEN + l] = bi;
    }
    __syncthreads();
    int kmin = kmin_s;
    if (threadIdx.x < CVQ_C) {
        float qv = codebook[(size_t)kmin * CVQ_C + threadIdx.x];
        q[((size_t)b * CVQ_C + threadIdx.x) * LLEN + l] = qv;
        float df = xs[threadIdx.x] - qv;
        atomicAdd(loss_accum, df * df);
    }
}

__global__ void vq_finalize(const float* __restrict__ accum, float* __restrict__ loss_out)
{
    // beta * mean over B*L*CVQ elements
    loss_out[0] = 0.25f * accum[0] * (1.0f / (float)(BATCH * LLEN * CVQ_C));
}

// ---------------------------------------------------------------------------
// Host-side orchestration
// ---------------------------------------------------------------------------
static inline void launch_gemm(hipStream_t s, const float* X, const float* W,
                               const float* bias, const float* resid, const float* mask,
                               float* Y, int B, int Cin, int Cout, int T,
                               float scale, int ksize, bool gelu)
{
    dim3 grid((T + TILE_N - 1) / TILE_N, (Cout + TILE_M - 1) / TILE_M, B);
    dim3 block(GEMM_THREADS);
    if (ksize == 3) {
        if (gelu) wmma_conv_gemm<3, true ><<<grid, block, 0, s>>>(X, W, bias, resid, mask, Y, Cin, Cout, T, scale);
        else      wmma_conv_gemm<3, false><<<grid, block, 0, s>>>(X, W, bias, resid, mask, Y, Cin, Cout, T, scale);
    } else {
        if (gelu) wmma_conv_gemm<1, true ><<<grid, block, 0, s>>>(X, W, bias, resid, mask, Y, Cin, Cout, T, scale);
        else      wmma_conv_gemm<1, false><<<grid, block, 0, s>>>(X, W, bias, resid, mask, Y, Cin, Cout, T, scale);
    }
}

struct BlockParams {
    const float *ln_g, *ln_b, *w1, *b1, *w2, *b2, *last_g, *last_b, *post_w, *post_b;
};

// Runs LeftPadConvBlocks; x lives in xbuf throughout; result left in xbuf.
static void run_conv_blocks(hipStream_t s, float* xbuf, float* tmp, float* mid,
                            const float* maskp, const BlockParams& P, int B, int T)
{
    const float scale1 = 0.57735026918962576f;   // KS^-0.5
    int ncols = B * T;
    dim3 lgrid((ncols + 7) / 8), lblk(256);
    for (int i = 0; i < NLAY; ++i) {
        for (int j = 0; j < NBLK; ++j) {
            int o = i * NBLK + j;
            ln_kernel<<<lgrid, lblk, 0, s>>>(xbuf, P.ln_g + (size_t)o * HC,
                                             P.ln_b + (size_t)o * HC, nullptr, tmp, T, ncols);
            launch_gemm(s, tmp, P.w1 + (size_t)o * H2 * HC * 3, P.b1 + (size_t)o * H2,
                        nullptr, nullptr, mid, B, HC, H2, T, scale1, 3, true);
            launch_gemm(s, mid, P.w2 + (size_t)o * HC * H2, P.b2 + (size_t)o * HC,
                        xbuf, maskp, xbuf, B, H2, HC, T, 1.0f, 1, false);
        }
    }
    // x = LN(x)*mask ; x = causal_conv(x)*mask
    ln_kernel<<<lgrid, lblk, 0, s>>>(xbuf, P.last_g, P.last_b, maskp, tmp, T, ncols);
    launch_gemm(s, tmp, P.post_w, P.post_b, nullptr, maskp, xbuf, B, HC, HC, T, 1.0f, 3, false);
}

extern "C" void kernel_launch(void* const* d_in, const int* in_sizes, int n_in,
                              void* d_out, int out_size, void* d_ws, size_t ws_size,
                              hipStream_t stream)
{
    (void)in_sizes; (void)n_in; (void)out_size; (void)ws_size;

    const float* x             = (const float*)d_in[0];
    const float* in_np         = (const float*)d_in[1];
    const int*   mel2ph        = (const int*)  d_in[2];
    const float* ph_np         = (const float*)d_in[3];
    const float* conv_in_w     = (const float*)d_in[4];
    const float* conv_in_b     = (const float*)d_in[5];

    BlockParams enc = {
        (const float*)d_in[6],  (const float*)d_in[7],  (const float*)d_in[8],
        (const float*)d_in[9],  (const float*)d_in[10], (const float*)d_in[11],
        (const float*)d_in[12], (const float*)d_in[13], (const float*)d_in[14],
        (const float*)d_in[15]
    };
    BlockParams pn = {
        (const float*)d_in[16], (const float*)d_in[17], (const float*)d_in[18],
        (const float*)d_in[19], (const float*)d_in[20], (const float*)d_in[21],
        (const float*)d_in[22], (const float*)d_in[23], (const float*)d_in[24],
        (const float*)d_in[25]
    };
    const float* proj_in_w  = (const float*)d_in[26];
    const float* proj_in_b  = (const float*)d_in[27];
    const float* proj_out_w = (const float*)d_in[28];
    const float* proj_out_b = (const float*)d_in[29];
    const float* codebook   = (const float*)d_in[30];

    // workspace layout (floats); regions are reused sequentially
    float* ws = (float*)d_ws;
    const size_t SZ_H = (size_t)BATCH * HC * TLEN;     // 8,388,608
    float* buf_h = ws;                                  // [B,H,T]
    float* buf_t = ws + SZ_H;                           // [B,H,T] (also grouped [B,H,L])
    float* buf_m = ws + 2 * SZ_H;                       // [B,2H,T] (reused below)
    float* sums   = buf_m;                              // [B,H,L+1]
    float* counts = sums + (size_t)BATCH * HC * (LLEN + 1);
    float* zbuf   = buf_m;                              // [B,CVQ,L] (after sums dead)
    float* qbuf   = buf_m + (size_t)BATCH * CVQ_C * LLEN;
    float* lossac = qbuf + (size_t)BATCH * CVQ_C * LLEN;

    // 1) conv_in: [B,80,T] -> [B,H,T], *mask
    launch_gemm(stream, x, conv_in_w, conv_in_b, nullptr, in_np, buf_h,
                BATCH, 80, HC, TLEN, 1.0f, 1, false);

    // 2) encoder conv blocks (result in buf_h)
    run_conv_blocks(stream, buf_h, buf_t, buf_m, in_np, enc, BATCH, TLEN);

    // 3) group mel frames by phoneme segments -> buf_t as [B,H,L]
    {
        size_t nz = (size_t)BATCH * HC * (LLEN + 1) + (size_t)BATCH * (LLEN + 1);
        zero_kernel<<<512, 256, 0, stream>>>(sums, nz);
        dim3 sg((TLEN * HC + 255) / 256, BATCH);
        scatter_kernel<<<sg, 256, 0, stream>>>(buf_h, mel2ph, sums, counts);
        dim3 ng((HC * LLEN + 255) / 256, BATCH);
        group_norm_kernel<<<ng, 256, 0, stream>>>(sums, counts, buf_t);
    }

    // 4) postnet conv blocks on [B,H,L] (result in buf_t)
    run_conv_blocks(stream, buf_t, buf_h, buf_m, ph_np, pn, BATCH, LLEN);

    // 5) proj_in: [B,H,L] -> z [B,CVQ,L]
    launch_gemm(stream, buf_t, proj_in_w, proj_in_b, nullptr, nullptr, zbuf,
                BATCH, HC, CVQ_C, LLEN, 1.0f, 1, false);

    // 6) VQ: nearest code, straight-through value q, commitment loss, idx
    float* out_f   = (float*)d_out;
    float* loss_o  = out_f + (size_t)BATCH * HC * LLEN;
    int*   idx_o   = (int*)(loss_o + 1);
    zero_kernel<<<1, 64, 0, stream>>>(lossac, 1);
    vq_kernel<<<BATCH * LLEN, KCODE, 0, stream>>>(zbuf, codebook, qbuf, idx_o, lossac);
    vq_finalize<<<1, 1, 0, stream>>>(lossac, loss_o);

    // 7) proj_out: q [B,CVQ,L] -> out [B,H,L]  (forward value of q_st is q)
    launch_gemm(stream, qbuf, proj_out_w, proj_out_b, nullptr, nullptr, out_f,
                BATCH, CVQ_C, HC, LLEN, 1.0f, 1, false);
}
